// SpatialClusteringCTGenerator_14224931684551
// MI455X (gfx1250) — matrix operand order repaired and checked by reading, not compiled, not run on hardware
//
#include <hip/hip_runtime.h>
#include <hip/hip_bf16.h>

// ---------------------------------------------------------------------------
// CDNA5 (gfx1250) wave32 WMMA pipeline for clustered-attention reference.
// bf16 inputs to v_wmma_f32_16x16x32_bf16, f32 accumulation.
// B-side operands pre-packed fragment-major ([tile][lane][16 bf16]) so every
// fragment load is 2 x b128. GEMM + pool are double-buffered and software-
// pipelined; A tiles use async global->LDS when the builtin is available.
// ---------------------------------------------------------------------------

#ifndef __has_builtin
#define __has_builtin(x) 0
#endif
#if __has_builtin(__builtin_amdgcn_global_load_async_to_lds_b128)
#define HAVE_ASYNC_LDS 1
#else
#define HAVE_ASYNC_LDS 0
#endif

typedef int v4i __attribute__((ext_vector_type(4)));

#if HAVE_ASYNC_LDS
__device__ __forceinline__ void async_lds_b128(const void* g, void* l) {
    // param0: v4i in AS(1) (prints as "int __vector(4) __device__ *")
    // param1: v4i in AS(3) (LDS destination)
    __builtin_amdgcn_global_load_async_to_lds_b128(
        (__attribute__((address_space(1))) v4i*)g,
        (__attribute__((address_space(3))) v4i*)l, 0, 0);
}
__device__ __forceinline__ void wait_async0() {
#if __has_builtin(__builtin_amdgcn_s_wait_asynccnt)
    __builtin_amdgcn_s_wait_asynccnt(0);
#else
    asm volatile("s_wait_asynccnt 0x0" ::: "memory");
#endif
}
#endif

typedef __bf16  v16bf __attribute__((ext_vector_type(16)));
typedef float   v8f   __attribute__((ext_vector_type(8)));

union Frag {                 // 16 bf16 values = 8 VGPRs per lane
    v16bf v;
    unsigned short u[16];
    uint4 q[2];
};

__device__ __forceinline__ unsigned short f32_to_bf16(float f) {
    unsigned int u = __builtin_bit_cast(unsigned int, f);
    u = (u + 0x7FFFu + ((u >> 16) & 1u)) >> 16;   // round-to-nearest-even
    return (unsigned short)u;
}

__device__ __forceinline__ v8f zero8() {
    v8f z = {0.f, 0.f, 0.f, 0.f, 0.f, 0.f, 0.f, 0.f};
    return z;
}

__device__ __forceinline__ v8f wmma_bf16(const Frag& a, const Frag& b, v8f c) {
    // D = A(16x32) * B(32x16) + C(16x16 f32)
    return __builtin_amdgcn_wmma_f32_16x16x32_bf16(
        false, a.v, false, b.v, (short)0, c, false, false);
}

// fragment-major packing helpers --------------------------------------------
// B tile (32 K x 16 cols): lane = ((k>>3)&1)*16 + col, slot = (k&7) + ((k>>4)<<3)
__device__ __forceinline__ int b_pack_idx(int k, int col15) {
    return (((k >> 3) & 1) * 16 + col15) * 16 + ((k & 7) + ((k >> 4) << 3));
}
// A tile (16 M x 32 K): lane = ((k>>3)&1)*16 + m, slot = (k&7) + ((k>>4)<<3)
__device__ __forceinline__ int a_pack_idx(int m15, int k) {
    return (((k >> 3) & 1) * 16 + m15) * 16 + ((k & 7) + ((k >> 4) << 3));
}
// scatter one uint4 (8 bf16 of row `row`, cols c4*8..+7) into packed B tiles
__device__ __forceinline__ void pack_b_tile(unsigned short* dst, const uint4& v,
                                            int row, int c4) {
    const unsigned short* e = (const unsigned short*)&v;
#pragma unroll
    for (int i = 0; i < 8; ++i) {
        int col = c4 * 8 + i;
        dst[(col >> 4) * 512 + b_pack_idx(row, col & 15)] = e[i];
    }
}
// scatter one uint4 (8 bf16: K-row `row`, M cols c4*8..+7) into packed A tiles
__device__ __forceinline__ void pack_a_tile(unsigned short* dst, const uint4& v,
                                            int row, int c4) {
    const unsigned short* e = (const unsigned short*)&v;
#pragma unroll
    for (int i = 0; i < 8; ++i) {
        int m = c4 * 8 + i;
        dst[(m >> 4) * 512 + a_pack_idx(m & 15, row)] = e[i];
    }
}

// problem constants
#define BB      2
#define NN      65536
#define CC      512
#define C3      1536
#define HEADS   8
#define HD      64
#define KCL     64
#define MTOT    (BB * NN)        // 131072 rows

// ---------------------------------------------------------------------------
// 0) zero f32 accumulators
// ---------------------------------------------------------------------------
__global__ void zero_kernel(float* p, int n) {
    int i = blockIdx.x * blockDim.x + threadIdx.x;
    int stride = gridDim.x * blockDim.x;
    for (; i < n; i += stride) p[i] = 0.f;
}

// ---------------------------------------------------------------------------
// 1) generic f32 -> bf16 convert
// ---------------------------------------------------------------------------
__global__ void cvt_bf16_kernel(const float* __restrict__ src,
                                unsigned short* __restrict__ dst, long n) {
    long i = (long)blockIdx.x * blockDim.x + threadIdx.x;
    long stride = (long)gridDim.x * blockDim.x;
    for (; i < n; i += stride) dst[i] = f32_to_bf16(src[i]);
}

// ---------------------------------------------------------------------------
// 2) cluster assignments: convert to bf16 + accumulate denom[b][k] = sum_n a
// ---------------------------------------------------------------------------
__global__ __launch_bounds__(256) void cvt_assign_kernel(
        const float* __restrict__ A, unsigned short* __restrict__ Abf,
        float* __restrict__ denom) {
    __shared__ float part[KCL];
    long base = (long)blockIdx.x * 4096;
    int b = (int)(base / ((long)NN * KCL));
    if (threadIdx.x < KCL) part[threadIdx.x] = 0.f;
    __syncthreads();
#pragma unroll
    for (int i = 0; i < 16; ++i) {
        long idx = base + (long)i * 256 + threadIdx.x;
        float v = A[idx];
        Abf[idx] = f32_to_bf16(v);
        atomicAdd(&part[(int)(idx & (KCL - 1))], v);   // k = idx % 64
    }
    __syncthreads();
    if (threadIdx.x < KCL)
        atomicAdd(&denom[b * KCL + threadIdx.x], part[threadIdx.x]);
}

// ---------------------------------------------------------------------------
// 3) tiled WMMA GEMM:  C(MxNt) = A(MxKd, bf16) * B(KdxNt, bf16) [+ bias]
//    block tile 128x64, 8 waves, wave tile 16x64.
//    Double-buffered LDS; A tile via async global->LDS, B tile permuted
//    through registers; next-tile loads overlap current-tile WMMAs.
// ---------------------------------------------------------------------------
__global__ __launch_bounds__(256) void wmma_gemm_kernel(
        const unsigned short* __restrict__ A,
        const unsigned short* __restrict__ Bm,
        const float* __restrict__ bias,      // may be null
        unsigned short* __restrict__ Cbf,    // bf16 out (or null)
        float* __restrict__ Cf,              // f32 out (or null)
        int Kd, int Nt) {
    __shared__ __align__(16) unsigned short lA[2][128 * 32];    // row-major
    __shared__ __align__(16) unsigned short lBp[2][2048];       // packed

    const int tid  = threadIdx.x;
    const int w    = tid >> 5;
    const int lane = tid & 31;
    const int r    = lane & 15;
    const int half = lane >> 4;
    const long m0  = (long)blockIdx.y * 128;
    const int  n0  = blockIdx.x * 64;

    const int rowA = tid >> 2, c4A = tid & 3;   // thread owns A uint4 tid, tid+256
    const int rowB = tid >> 3, c4B = tid & 7;

    v8f acc[4];
#pragma unroll
    for (int i = 0; i < 4; ++i) acc[i] = zero8();

    // ---- prologue: stage k0 = 0 into buffer 0
    {
        const unsigned short* sA0 = A + (m0 + rowA) * (long)Kd + c4A * 8;
        const unsigned short* sA1 = A + (m0 + rowA + 64) * (long)Kd + c4A * 8;
#if HAVE_ASYNC_LDS
        async_lds_b128(sA0, &lA[0][tid * 8]);
        async_lds_b128(sA1, &lA[0][(tid + 256) * 8]);
#else
        ((uint4*)lA[0])[tid]       = *(const uint4*)sA0;
        ((uint4*)lA[0])[tid + 256] = *(const uint4*)sA1;
#endif
        uint4 b4 = *(const uint4*)(Bm + (long)rowB * Nt + n0 + c4B * 8);
        pack_b_tile(lBp[0], b4, rowB, c4B);
#if HAVE_ASYNC_LDS
        wait_async0();
#endif
    }
    __syncthreads();

    int cur = 0;
    for (int k0 = 0; k0 < Kd; k0 += 32) {
        const int nxt = k0 + 32;
        const int nb  = cur ^ 1;
        const bool more = nxt < Kd;
        uint4 bStage;
#if !HAVE_ASYNC_LDS
        uint4 aStage0, aStage1;
#endif
        if (more) {   // issue next-tile loads (overlap with WMMA below)
#if HAVE_ASYNC_LDS
            async_lds_b128(A + (m0 + rowA) * (long)Kd + nxt + c4A * 8,
                           &lA[nb][tid * 8]);
            async_lds_b128(A + (m0 + rowA + 64) * (long)Kd + nxt + c4A * 8,
                           &lA[nb][(tid + 256) * 8]);
#else
            aStage0 = *(const uint4*)(A + (m0 + rowA) * (long)Kd + nxt + c4A * 8);
            aStage1 = *(const uint4*)(A + (m0 + rowA + 64) * (long)Kd + nxt + c4A * 8);
#endif
            bStage = *(const uint4*)(Bm + (long)(nxt + rowB) * Nt + n0 + c4B * 8);
            if (nxt + 32 < Kd)   // hint the tile after next
                __builtin_prefetch(Bm + (long)(nxt + 32 + rowB) * Nt + n0, 0, 0);
        }

        // compute current tile
        Frag a;
        a.q[0] = ((const uint4*)lA[cur])[(w * 16 + r) * 4 + half];
        a.q[1] = ((const uint4*)lA[cur])[(w * 16 + r) * 4 + 2 + half];
#pragma unroll
        for (int nt = 0; nt < 4; ++nt) {
            Frag b;
            b.q[0] = *(const uint4*)&lBp[cur][nt * 512 + lane * 16];
            b.q[1] = *(const uint4*)&lBp[cur][nt * 512 + lane * 16 + 8];
            acc[nt] = wmma_bf16(a, b, acc[nt]);
        }

        if (more) {   // commit next tile
            pack_b_tile(lBp[nb], bStage, rowB, c4B);
#if HAVE_ASYNC_LDS
            wait_async0();
#else
            ((uint4*)lA[nb])[tid]       = aStage0;
            ((uint4*)lA[nb])[tid + 256] = aStage1;
#endif
            __syncthreads();
        }
        cur = nb;
    }

    // store: C layout -> VGPR rr holds row (half*8+rr), col = nt*16 + r
#pragma unroll
    for (int nt = 0; nt < 4; ++nt) {
#pragma unroll
        for (int rr = 0; rr < 8; ++rr) {
            long row = m0 + w * 16 + half * 8 + rr;
            int  col = n0 + nt * 16 + r;
            float v = acc[nt][rr];
            if (bias) v += bias[col];
            if (Cf)   Cf[row * (long)Nt + col] = v;
            else      Cbf[row * (long)Nt + col] = f32_to_bf16(v);
        }
    }
}

// ---------------------------------------------------------------------------
// 4) cluster pooling, split-K over N, double-buffered packed LDS staging:
//    kpool[b][kc][c] += sum_n a[b][n][kc] * K[b][n][c]   (same for V)
//    grid = (8 col-blocks of 64, 32 N-splits of 2048, B)
// ---------------------------------------------------------------------------
__global__ __launch_bounds__(256) void pool_kernel(
        const unsigned short* __restrict__ Abf,     // (B*N, 64)
        const unsigned short* __restrict__ qkvbf,   // (B*N, 1536)
        float* __restrict__ kpool, float* __restrict__ vpool) {
    __shared__ __align__(16) unsigned short aP[2][2048];   // A^T packed
    __shared__ __align__(16) unsigned short kP[2][2048];
    __shared__ __align__(16) unsigned short vP[2][2048];

    const int b    = blockIdx.z;
    const int n0c  = blockIdx.x * 64;                  // C column offset
    const long rowbase = (long)b * NN + (long)blockIdx.y * 2048;

    const int tid  = threadIdx.x;
    const int w    = tid >> 5;
    const int lane = tid & 31;
    const int r    = lane & 15;
    const int half = lane >> 4;
    const int mt   = w >> 1;            // cluster tile 0..3
    const int ntp  = (w & 1) * 2;       // n-tile pair
    const int row  = tid >> 3, c4 = tid & 7;   // staging coords (row = local n)

    v8f ak[2], av[2];
    ak[0] = zero8(); ak[1] = zero8(); av[0] = zero8(); av[1] = zero8();

    // prologue: stage chunk 0 into buffer 0
    {
        uint4 a4 = *(const uint4*)(Abf + (rowbase + row) * KCL + c4 * 8);
        long gr = (rowbase + row) * C3 + n0c + c4 * 8;
        uint4 k4 = *(const uint4*)(qkvbf + gr + CC);
        uint4 v4 = *(const uint4*)(qkvbf + gr + 2 * CC);
        pack_a_tile(aP[0], a4, row, c4);
        pack_b_tile(kP[0], k4, row, c4);
        pack_b_tile(vP[0], v4, row, c4);
    }
    __syncthreads();

    int cur = 0;
    for (int kk = 0; kk < 2048; kk += 32) {
        const int nb = cur ^ 1;
        const bool more = kk + 32 < 2048;
        uint4 a4, k4, v4;
        if (more) {   // issue next-chunk loads (overlap with WMMA below)
            a4 = *(const uint4*)(Abf + (rowbase + kk + 32 + row) * KCL + c4 * 8);
            long gr = (rowbase + kk + 32 + row) * C3 + n0c + c4 * 8;
            k4 = *(const uint4*)(qkvbf + gr + CC);
            v4 = *(const uint4*)(qkvbf + gr + 2 * CC);
        }

        Frag a;
        a.q[0] = *(const uint4*)&aP[cur][mt * 512 + lane * 16];
        a.q[1] = *(const uint4*)&aP[cur][mt * 512 + lane * 16 + 8];
#pragma unroll
        for (int t = 0; t < 2; ++t) {
            const int nt = ntp + t;
            Frag bk, bv;
            bk.q[0] = *(const uint4*)&kP[cur][nt * 512 + lane * 16];
            bk.q[1] = *(const uint4*)&kP[cur][nt * 512 + lane * 16 + 8];
            bv.q[0] = *(const uint4*)&vP[cur][nt * 512 + lane * 16];
            bv.q[1] = *(const uint4*)&vP[cur][nt * 512 + lane * 16 + 8];
            ak[t] = wmma_bf16(a, bk, ak[t]);
            av[t] = wmma_bf16(a, bv, av[t]);
        }

        if (more) {
            pack_a_tile(aP[nb], a4, row, c4);
            pack_b_tile(kP[nb], k4, row, c4);
            pack_b_tile(vP[nb], v4, row, c4);
            __syncthreads();
        }
        cur = nb;
    }
#pragma unroll
    for (int t = 0; t < 2; ++t) {
#pragma unroll
        for (int rr = 0; rr < 8; ++rr) {
            int krow = mt * 16 + half * 8 + rr;
            int col  = n0c + (ntp + t) * 16 + r;
            long o = ((long)b * KCL + krow) * CC + col;
            atomicAdd(&kpool[o], ak[t][rr]);
            atomicAdd(&vpool[o], av[t][rr]);
        }
    }
}

// ---------------------------------------------------------------------------
// 5) normalize pools by (denom + eps); emit kc/vc directly in fragment-packed
//    GLOBAL layout so the attention kernel loads B-frags as contiguous b128:
//    kcP[b][h][ks][nt][lane][16]: logits B-tile  (K-dim = d, cols = clusters)
//    vcP[b][h][ks][nt][lane][16]: output B-tile  (K-dim = clusters, cols = d)
// ---------------------------------------------------------------------------
__global__ void finalize_pool_kernel(
        const float* __restrict__ kpool, const float* __restrict__ vpool,
        const float* __restrict__ denom,
        unsigned short* __restrict__ kcP, unsigned short* __restrict__ vcP) {
    int idx = blockIdx.x * blockDim.x + threadIdx.x;   // (b, kcl, cfull)
    int cfull = idx & (CC - 1);
    int kcl   = (idx >> 9) & (KCL - 1);
    int b     = idx >> 15;
    float inv = 1.0f / (denom[b * KCL + kcl] + 1e-8f);
    unsigned short kv = f32_to_bf16(kpool[idx] * inv);
    unsigned short vv = f32_to_bf16(vpool[idx] * inv);
    int h = cfull >> 6, d = cfull & (HD - 1);
    {   // kc: tile(ks = d/32, nt = kcl/16), element (k = d%32, col = kcl%16)
        int ks = d >> 5, k = d & 31;
        int nt = kcl >> 4, c = kcl & 15;
        kcP[(((b * HEADS + h) * 2 + ks) * 4 + nt) * 512 + b_pack_idx(k, c)] = kv;
    }
    {   // vc: tile(ks = kcl/32, nt = d/16), element (k = kcl%32, col = d%16)
        int ks = kcl >> 5, k = kcl & 31;
        int nt = d >> 4, c = d & 15;
        vcP[(((b * HEADS + h) * 2 + ks) * 4 + nt) * 512 + b_pack_idx(k, c)] = vv;
    }
}

// cluster_bias (64x64 f32) -> fragment-packed bf16: biasP[ks][nt][lane][16]
__global__ void pack_bias_kernel(const float* __restrict__ cbias,
                                 unsigned short* __restrict__ biasP) {
    int idx = blockIdx.x * blockDim.x + threadIdx.x;   // 4096
    int c = idx & (KCL - 1), k = idx >> 6;
    int ks = k >> 5, kl = k & 31, nt = c >> 4, cl = c & 15;
    biasP[(ks * 4 + nt) * 512 + b_pack_idx(kl, cl)] = f32_to_bf16(cbias[idx]);
}

// ---------------------------------------------------------------------------
// 6) fused attention: wave = head, block = 64 voxels.
//    logits = q . k_c^T * hd^-0.5 + a . cluster_bias ; softmax over 64
//    x = attn . v_c   (all matmuls on WMMA, all B-frags packed b128 loads)
// ---------------------------------------------------------------------------
__global__ __launch_bounds__(256) void attn_kernel(
        const unsigned short* __restrict__ Abf,     // (B*N, 64)
        const unsigned short* __restrict__ qkvbf,   // (B*N, 1536)
        const unsigned short* __restrict__ kcP,     // packed
        const unsigned short* __restrict__ vcP,     // packed
        const unsigned short* __restrict__ biasP,   // packed
        unsigned short* __restrict__ xout) {        // (B*N, 512)
    __shared__ __align__(16) unsigned short attn_lds[HEADS][16 * 64];

    const int b    = blockIdx.y;
    const long nbase = (long)blockIdx.x * 64;
    const int h    = threadIdx.x >> 5;     // wave index == head
    const int lane = threadIdx.x & 31;
    const int r    = lane & 15;
    const int half = lane >> 4;
    const float scale = 0.125f;            // hd^-0.5 = 1/8

    const unsigned short* kcH = kcP + (long)(b * HEADS + h) * 4096;
    const unsigned short* vcH = vcP + (long)(b * HEADS + h) * 4096;

    for (int mt = 0; mt < 4; ++mt) {
        const long rowv = (long)b * NN + nbase + mt * 16;   // first voxel row

        v8f accq[4], accb[4];
#pragma unroll
        for (int i = 0; i < 4; ++i) { accq[i] = zero8(); accb[i] = zero8(); }

#pragma unroll
        for (int ks = 0; ks < 2; ++ks) {
            const int k0 = ks * 32;
            Frag aq, aa;
            {   // A-frags: contiguous 16B pairs from global rows
                long rw = rowv + r;
                const unsigned short* ap = Abf + rw * KCL + k0 + half * 8;
                aa.q[0] = *(const uint4*)ap;
                aa.q[1] = *(const uint4*)(ap + 16);
                const unsigned short* qp = qkvbf + rw * C3 + h * HD + k0 + half * 8;
                aq.q[0] = *(const uint4*)qp;
                aq.q[1] = *(const uint4*)(qp + 16);
            }
#pragma unroll
            for (int nt = 0; nt < 4; ++nt) {
                Frag bb, bq;
                const unsigned short* pb = biasP + (ks * 4 + nt) * 512 + lane * 16;
                bb.q[0] = *(const uint4*)pb;
                bb.q[1] = *(const uint4*)(pb + 8);
                const unsigned short* pk = kcH + (ks * 4 + nt) * 512 + lane * 16;
                bq.q[0] = *(const uint4*)pk;
                bq.q[1] = *(const uint4*)(pk + 8);
                accb[nt] = wmma_bf16(aa, bb, accb[nt]);
                accq[nt] = wmma_bf16(aq, bq, accq[nt]);
            }
        }

        // row-wise softmax over 64 clusters. In the C layout a row lives in
        // one VGPR slot across the 16 lanes of a half + 4 n-tile registers.
        float lg[4][8];
#pragma unroll
        for (int rr = 0; rr < 8; ++rr) {
            float m = -3.4e38f;
#pragma unroll
            for (int nt = 0; nt < 4; ++nt) {
                lg[nt][rr] = accq[nt][rr] * scale + accb[nt][rr];
                m = fmaxf(m, lg[nt][rr]);
            }
#pragma unroll
            for (int d = 1; d < 16; d <<= 1) m = fmaxf(m, __shfl_xor(m, d, 32));
            float s = 0.f;
#pragma unroll
            for (int nt = 0; nt < 4; ++nt) {
                float e = __expf(lg[nt][rr] - m);
                lg[nt][rr] = e;
                s += e;
            }
#pragma unroll
            for (int d = 1; d < 16; d <<= 1) s += __shfl_xor(s, d, 32);
            float inv = 1.0f / s;
#pragma unroll
            for (int nt = 0; nt < 4; ++nt)
                attn_lds[h][(half * 8 + rr) * 64 + nt * 16 + r] =
                    f32_to_bf16(lg[nt][rr] * inv);
        }
        __syncthreads();   // LDS round-trip: reshape C layout -> A fragment

        v8f accx[4];
#pragma unroll
        for (int i = 0; i < 4; ++i) accx[i] = zero8();
#pragma unroll
        for (int ks = 0; ks < 2; ++ks) {
            const int k0 = ks * 32;
            Frag aa;
            const unsigned short* ap = &attn_lds[h][r * 64 + k0 + half * 8];
            aa.q[0] = *(const uint4*)ap;
            aa.q[1] = *(const uint4*)(ap + 16);
#pragma unroll
            for (int nt = 0; nt < 4; ++nt) {
                Frag bv;
                const unsigned short* pv = vcH + (ks * 4 + nt) * 512 + lane * 16;
                bv.q[0] = *(const uint4*)pv;
                bv.q[1] = *(const uint4*)(pv + 8);
                accx[nt] = wmma_bf16(aa, bv, accx[nt]);
            }
        }
#pragma unroll
        for (int nt = 0; nt < 4; ++nt)
#pragma unroll
            for (int rr = 0; rr < 8; ++rr)
                xout[(rowv + half * 8 + rr) * CC + h * HD + nt * 16 + r] =
                    f32_to_bf16(accx[nt][rr]);
        __syncthreads();
    }
}

// ---------------------------------------------------------------------------
// launch
// ---------------------------------------------------------------------------
extern "C" void kernel_launch(void* const* d_in, const int* in_sizes, int n_in,
                              void* d_out, int out_size, void* d_ws, size_t ws_size,
                              hipStream_t stream) {
    const float* vf    = (const float*)d_in[0];   // (B,N,C)
    const float* ca    = (const float*)d_in[1];   // (B,N,K)
    const float* wqkv  = (const float*)d_in[2];   // (C,3C)
    const float* wproj = (const float*)d_in[3];   // (C,C)
    const float* bproj = (const float*)d_in[4];   // (C,)
    const float* cbias = (const float*)d_in[5];   // (K,K)

    size_t off = 0;
    char* base = (char*)d_ws;
    auto take = [&](size_t bytes) -> char* {
        char* p = base + off;
        off += (bytes + 255) & ~(size_t)255;
        return p;
    };
    // xbuf: holds X bf16 for QKV GEMM, then re-used as attention output x
    unsigned short* xbuf     = (unsigned short*)take((size_t)MTOT * CC * 2);
    unsigned short* abf      = (unsigned short*)take((size_t)MTOT * KCL * 2);
    unsigned short* wqkv_bf  = (unsigned short*)take((size_t)CC * C3 * 2);
    unsigned short* wproj_bf = (unsigned short*)take((size_t)CC * CC * 2);
    unsigned short* biasP    = (unsigned short*)take((size_t)2 * 4 * 512 * 2);
    unsigned short* qkv_bf   = (unsigned short*)take((size_t)MTOT * C3 * 2);
    float* denom = (float*)take((size_t)(BB * KCL + 2 * BB * KCL * CC) * 4);
    float* kpool = denom + BB * KCL;
    float* vpool = kpool + BB * KCL * CC;
    unsigned short* kcP = (unsigned short*)take((size_t)BB * KCL * CC * 2);
    unsigned short* vcP = (unsigned short*)take((size_t)BB * KCL * CC * 2);

    // 0) zero accumulators (denom + kpool + vpool, contiguous)
    zero_kernel<<<256, 256, 0, stream>>>(denom, BB * KCL + 2 * BB * KCL * CC);
    // 1) convert inputs / weights to bf16 (+pack cluster_bias)
    cvt_bf16_kernel<<<2048, 256, 0, stream>>>(vf, xbuf, (long)MTOT * CC);
    cvt_assign_kernel<<<(MTOT * KCL) / 4096, 256, 0, stream>>>(ca, abf, denom);
    cvt_bf16_kernel<<<256, 256, 0, stream>>>(wqkv, wqkv_bf, (long)CC * C3);
    cvt_bf16_kernel<<<256, 256, 0, stream>>>(wproj, wproj_bf, (long)CC * CC);
    pack_bias_kernel<<<16, 256, 0, stream>>>(cbias, biasP);
    // 2) QKV GEMM: (131072x512)*(512x1536) -> bf16
    wmma_gemm_kernel<<<dim3(C3 / 64, MTOT / 128), 256, 0, stream>>>(
        xbuf, wqkv_bf, nullptr, qkv_bf, nullptr, CC, C3);
    // 3) cluster pooling (split-K, f32 atomic accumulate)
    pool_kernel<<<dim3(CC / 64, NN / 2048, BB), 256, 0, stream>>>(
        abf, qkv_bf, kpool, vpool);
    // 4) normalize pools -> packed kc/vc
    finalize_pool_kernel<<<(BB * KCL * CC) / 256, 256, 0, stream>>>(
        kpool, vpool, denom, kcP, vcP);
    // 5) fused attention -> xbuf (bf16)
    attn_kernel<<<dim3(NN / 64, BB), 256, 0, stream>>>(
        abf, qkv_bf, kcP, vcP, biasP, xbuf);
    // 6) output projection: (131072x512)*(512x512) + b -> f32 d_out
    wmma_gemm_kernel<<<dim3(CC / 64, MTOT / 128), 256, 0, stream>>>(
        xbuf, wproj_bf, bproj, nullptr, (float*)d_out, CC, CC);
}
